// WindowAttention_11639361372227
// MI455X (gfx1250) — matrix operand order repaired and checked by reading, not compiled, not run on hardware
//
#include <hip/hip_runtime.h>
#include <hip/hip_bf16.h>

typedef __attribute__((ext_vector_type(16))) _Float16 v16h;
typedef __attribute__((ext_vector_type(8)))  float    v8f;

union F16x16 { v16h v; uint4 u[2]; };
union H2 { _Float16 h[2]; unsigned u; };

__device__ inline v16h ld2(const _Float16* p0, const _Float16* p1) {
  F16x16 f;
  f.u[0] = *(const uint4*)p0;
  f.u[1] = *(const uint4*)p1;
  return f.v;
}

// Low 32 bits of a generic pointer to __shared__ == LDS byte offset (flat aperture rule).
__device__ inline unsigned lds_off(const void* p) {
  return (unsigned)(unsigned long long)p;
}

#define WMMA_F16(A, B, C) \
  __builtin_amdgcn_wmma_f32_16x16x32_f16(false, (A), false, (B), (short)0, (C), false, false)

// ---------------------------------------------------------------------------
// Kernel 1: CPB-MLP relative position bias -> biasw[h][i][j] = 16*sigmoid(rpb)
// ---------------------------------------------------------------------------
__global__ __launch_bounds__(256) void cpb_bias_kernel(
    const float* __restrict__ w1, const float* __restrict__ b1,
    const float* __restrict__ w2, const float* __restrict__ b2,
    float* __restrict__ biasw) {
  __shared__ float tbl[225];
  const int h = blockIdx.x;
  const int tid = threadIdx.x;
  if (tid < 225) {
    int i = tid / 15, j = tid % 15;
    float v0 = (float)(i - 7) * (8.0f / 7.0f);
    float v1 = (float)(j - 7) * (8.0f / 7.0f);
    float s0 = (v0 > 0.f) ? 1.f : ((v0 < 0.f) ? -1.f : 0.f);
    float s1 = (v1 > 0.f) ? 1.f : ((v1 < 0.f) ? -1.f : 0.f);
    float c0 = s0 * log2f(fabsf(v0) + 1.f) * (1.0f / 3.0f);
    float c1 = s1 * log2f(fabsf(v1) + 1.f) * (1.0f / 3.0f);
    float out = b2[h];
    for (int u = 0; u < 64; u++) {
      float hid = fmaxf(c0 * w1[u] + c1 * w1[64 + u] + b1[u], 0.f);
      out += hid * w2[u * 16 + h];
    }
    tbl[tid] = out;
  }
  __syncthreads();
  for (int p = tid; p < 4096; p += 256) {
    int i = p >> 6, j = p & 63;
    int ry = (i >> 3) - (j >> 3) + 7;
    int rx = (i & 7) - (j & 7) + 7;
    float v = tbl[ry * 15 + rx];
    biasw[h * 4096 + p] = 16.0f / (1.0f + __expf(-v));
  }
}

// ---------------------------------------------------------------------------
// Kernel 2: QKV projection. BM=64, BN=128, BK=32. A = roll(x) fused gather.
// Each wave: 16x128 strip. All frags loaded first, then 8 back-to-back WMMAs.
// ---------------------------------------------------------------------------
__global__ __launch_bounds__(128) void gemm_qkv_kernel(
    const float* __restrict__ x, const float* __restrict__ wqkv,
    float* __restrict__ qw, float* __restrict__ kw, float* __restrict__ vw) {
  __shared__ __align__(16) _Float16 Ash[64][40];
  __shared__ __align__(16) _Float16 Bsh[128][40];   // transposed: [col][k]
  const int tid = threadIdx.x;
  const int lane = tid & 31, wv = tid >> 5;
  const int g = lane >> 4, n = lane & 15;
  const int m0 = blockIdx.x * 64, n0 = blockIdx.y * 128;

  v8f acc[8];
  #pragma unroll
  for (int t = 0; t < 8; t++)
    #pragma unroll
    for (int e = 0; e < 8; e++) acc[t][e] = 0.f;

  // A loader: row = tid>>1 (0..63), 16 cols at (tid&1)*16; fused +4 cyclic roll
  const int arow = tid >> 1, ac0 = (tid & 1) * 16;
  const int grow = m0 + arow;
  const int ab = grow >> 12, ay = (grow >> 6) & 63, ax = grow & 63;
  const float* aprow =
      x + (((size_t)ab * 64 + ((ay + 4) & 63)) * 64 + ((ax + 4) & 63)) * 512;
  // B loader: k-pair = tid>>3 (0..15), 16 cols at (tid&7)*16, packed b32 stores
  const int kp = tid >> 3, bc0 = (tid & 7) * 16;

  for (int k0 = 0; k0 < 512; k0 += 32) {
    __syncthreads();
    const float* ap = aprow + k0 + ac0;
    #pragma unroll
    for (int j = 0; j < 16; j++) Ash[arow][ac0 + j] = (_Float16)ap[j];
    const float* bp0 = wqkv + (size_t)(k0 + 2 * kp) * 1536 + n0 + bc0;
    const float* bp1 = bp0 + 1536;
    #pragma unroll
    for (int j = 0; j < 16; j++) {
      H2 pk;
      pk.h[0] = (_Float16)bp0[j];
      pk.h[1] = (_Float16)bp1[j];
      *(unsigned*)&Bsh[bc0 + j][2 * kp] = pk.u;
    }
    if (k0 < 480) {                       // prefetch next K-step tiles
      __builtin_prefetch(ap + 32);
      __builtin_prefetch(bp0 + 32 * 1536);
    }
    __syncthreads();
    // batch all frag loads -> one DS drain -> 8 consecutive WMMAs
    v16h a = ld2(&Ash[wv * 16 + n][g * 8], &Ash[wv * 16 + n][16 + g * 8]);
    v16h bf[8];
    #pragma unroll
    for (int nt = 0; nt < 8; nt++)
      bf[nt] = ld2(&Bsh[nt * 16 + n][g * 8], &Bsh[nt * 16 + n][16 + g * 8]);
    #pragma unroll
    for (int nt = 0; nt < 8; nt++)
      acc[nt] = WMMA_F16(a, bf[nt], acc[nt]);
  }

  // epilogue: scatter into windowed q/k/v layout [b, win, head][token][d]
  #pragma unroll
  for (int nt = 0; nt < 8; nt++) {
    int gcol = n0 + nt * 16 + n;
    int part = gcol >> 9, cc = gcol & 511;
    int hh = cc >> 5, dd = cc & 31;
    float* dst = (part == 0) ? qw : (part == 1) ? kw : vw;
    #pragma unroll
    for (int r = 0; r < 8; r++) {
      int row = m0 + wv * 16 + r + 8 * g;
      int b = row >> 12, yy = (row >> 6) & 63, xx = row & 63;
      int wh = yy >> 3, ww = xx >> 3, t = ((yy & 7) << 3) | (xx & 7);
      size_t idx = ((((size_t)((b << 6) + (wh << 3) + ww)) * 16 + hh) * 64 + t) * 32 + dd;
      dst[idx] = acc[nt][r];
    }
  }
}

// ---------------------------------------------------------------------------
// Kernel 3: windowed cosine attention, one block per (b, window, head).
// ---------------------------------------------------------------------------
__global__ __launch_bounds__(128) void attn_kernel(
    const float* __restrict__ qw, const float* __restrict__ kw,
    const float* __restrict__ vw, const float* __restrict__ logit_scale,
    const float* __restrict__ biasw, _Float16* __restrict__ att) {
  __shared__ __align__(16) _Float16 qsh[64][40];
  __shared__ __align__(16) _Float16 ksh[64][40];
  __shared__ __align__(16) _Float16 vsh[32][72];   // transposed: [d][token]
  __shared__ __align__(16) _Float16 Psh[64][72];

  const int bid = blockIdx.x;                      // ((b*64 + w)*16 + h)
  const int h = bid & 15, w = (bid >> 4) & 63, b = bid >> 10;
  const int wh = w >> 3, ww = w & 7;
  const int tid = threadIdx.x, lane = tid & 31, wv = tid >> 5;
  const int g = lane >> 4, n = lane & 15, rm0 = wv * 16;

  const size_t blk = (size_t)bid * 2048;
  const float* qb = qw + blk;
  const float* kb = kw + blk;
  const float* vb = vw + blk;

  // L2-normalize q rows (threads 0-63) / k rows (threads 64-127) -> f16 LDS
  {
    int row = tid & 63;
    const float* src = (tid < 64) ? (qb + row * 32) : (kb + row * 32);
    float tmp[32]; float ss = 0.f;
    #pragma unroll
    for (int d = 0; d < 32; d++) { tmp[d] = src[d]; ss += tmp[d] * tmp[d]; }
    float inv = 1.f / fmaxf(sqrtf(ss), 1e-12f);
    _Float16 (*dst)[40] = (tid < 64) ? qsh : ksh;
    #pragma unroll
    for (int d = 0; d < 32; d++) dst[row][d] = (_Float16)(tmp[d] * inv);
  }
  // v -> LDS transposed
  {
    int rv = tid >> 1, c0 = (tid & 1) * 16;
    const float* src = vb + rv * 32 + c0;
    #pragma unroll
    for (int d = 0; d < 16; d++) vsh[c0 + d][rv] = (_Float16)src[d];
  }
  __syncthreads();

  const float sc = __expf(fminf(logit_scale[h], 4.6051702f)); // log(100)
  const float* bh = biasw + h * 4096;

  // S = qn * kn^T  (each wave: 16x64 strip; frags batched, 4 WMMAs back-to-back)
  v8f s[4];
  #pragma unroll
  for (int t = 0; t < 4; t++)
    #pragma unroll
    for (int e = 0; e < 8; e++) s[t][e] = 0.f;
  {
    v16h a = ld2(&qsh[rm0 + n][g * 8], &qsh[rm0 + n][16 + g * 8]);
    v16h bf[4];
    #pragma unroll
    for (int nt = 0; nt < 4; nt++)
      bf[nt] = ld2(&ksh[nt * 16 + n][g * 8], &ksh[nt * 16 + n][16 + g * 8]);
    #pragma unroll
    for (int nt = 0; nt < 4; nt++)
      s[nt] = WMMA_F16(a, bf[nt], s[nt]);
  }
  // scale + CPB bias + analytic shift masks
  #pragma unroll
  for (int nt = 0; nt < 4; nt++) {
    int j = nt * 16 + n;
    #pragma unroll
    for (int r = 0; r < 8; r++) {
      int i = rm0 + r + 8 * g;
      float v = s[nt][r] * sc + bh[i * 64 + j];
      bool mUL = (wh == 7) && ((i >= 32) != (j >= 32));
      bool mLR = (ww == 7) && (((i & 7) >= 4) != ((j & 7) >= 4));
      s[nt][r] = (mUL || mLR) ? -1e30f : v;
    }
  }
  // row softmax: row m = rm0 + r + 8g lives across 16 lanes of the same half
  #pragma unroll
  for (int r = 0; r < 8; r++) {
    float mx = fmaxf(fmaxf(s[0][r], s[1][r]), fmaxf(s[2][r], s[3][r]));
    for (int off = 1; off < 16; off <<= 1) mx = fmaxf(mx, __shfl_xor(mx, off, 32));
    float sum = 0.f;
    #pragma unroll
    for (int nt = 0; nt < 4; nt++) { float e = __expf(s[nt][r] - mx); s[nt][r] = e; sum += e; }
    for (int off = 1; off < 16; off <<= 1) sum += __shfl_xor(sum, off, 32);
    float inv = 1.f / sum;
    #pragma unroll
    for (int nt = 0; nt < 4; nt++) s[nt][r] *= inv;
  }
  // P -> LDS (f16)
  #pragma unroll
  for (int nt = 0; nt < 4; nt++)
    #pragma unroll
    for (int r = 0; r < 8; r++)
      Psh[rm0 + r + 8 * g][nt * 16 + n] = (_Float16)s[nt][r];
  __syncthreads();

  // O = P * v  (each wave: 16x32; all frags batched, 4 WMMAs back-to-back)
  v8f o[2];
  #pragma unroll
  for (int t = 0; t < 2; t++)
    #pragma unroll
    for (int e = 0; e < 8; e++) o[t][e] = 0.f;
  {
    v16h pa[2], vf[4];
    #pragma unroll
    for (int kk = 0; kk < 2; kk++) {
      int ks = kk * 32;
      pa[kk] = ld2(&Psh[rm0 + n][ks + g * 8], &Psh[rm0 + n][ks + 16 + g * 8]);
      #pragma unroll
      for (int ot = 0; ot < 2; ot++)
        vf[kk * 2 + ot] =
            ld2(&vsh[ot * 16 + n][ks + g * 8], &vsh[ot * 16 + n][ks + 16 + g * 8]);
    }
    #pragma unroll
    for (int kk = 0; kk < 2; kk++)
      #pragma unroll
      for (int ot = 0; ot < 2; ot++)
        o[ot] = WMMA_F16(pa[kk], vf[kk * 2 + ot], o[ot]);
  }
  // store to de-windowed [b, y, x, h*32+d] (still shifted coords), f16
  #pragma unroll
  for (int ot = 0; ot < 2; ot++) {
    #pragma unroll
    for (int r = 0; r < 8; r++) {
      int t = rm0 + r + 8 * g;
      int y = wh * 8 + (t >> 3), xq = ww * 8 + (t & 7);
      int col = h * 32 + ot * 16 + n;
      att[(((size_t)b * 64 + y) * 64 + xq) * 512 + col] = (_Float16)o[ot][r];
    }
  }
}

// ---------------------------------------------------------------------------
// Kernel 4: out projection + bias, roll-back fused into the store index.
// BM=64, BN=128. A tile (already f16) staged via ASYNC global->LDS copies.
// ---------------------------------------------------------------------------
__global__ __launch_bounds__(128) void gemm_out_kernel(
    const _Float16* __restrict__ att, const float* __restrict__ wout,
    const float* __restrict__ bout, float* __restrict__ out) {
  __shared__ __align__(16) _Float16 Ash[64][40];
  __shared__ __align__(16) _Float16 Bsh[128][40];   // transposed: [col][k]
  const int tid = threadIdx.x;
  const int lane = tid & 31, wv = tid >> 5;
  const int g = lane >> 4, n = lane & 15;
  const int m0 = blockIdx.x * 64, n0 = blockIdx.y * 128;

  v8f acc[8];
  #pragma unroll
  for (int t = 0; t < 8; t++)
    #pragma unroll
    for (int e = 0; e < 8; e++) acc[t][e] = 0.f;

  const int arow = tid >> 1, ac0 = (tid & 1) * 16;
  const int kp = tid >> 3, bc0 = (tid & 7) * 16;
  const unsigned ldsA = lds_off(&Ash[arow][ac0]);

  for (int k0 = 0; k0 < 512; k0 += 32) {
    __syncthreads();
    // A tile: f16 -> f16 pure copy via CDNA5 async global->LDS DMA.
    // inst_offset applies to both LDS and global addresses, so two b128 ops
    // with offset 0 / 16 move the lane's 32 contiguous bytes.
    {
      unsigned voff = (unsigned)(((m0 + arow) * 512 + k0 + ac0) * 2);
      asm volatile(
          "global_load_async_to_lds_b128 %0, %1, %2\n\t"
          "global_load_async_to_lds_b128 %0, %1, %2 offset:16"
          :: "v"(ldsA), "v"(voff), "s"(att) : "memory");
    }
    // B tile: fp32 -> f16 convert, packed b32 transposed stores
    const float* bp0 = wout + (size_t)(k0 + 2 * kp) * 512 + n0 + bc0;
    const float* bp1 = bp0 + 512;
    #pragma unroll
    for (int j = 0; j < 16; j++) {
      H2 pk;
      pk.h[0] = (_Float16)bp0[j];
      pk.h[1] = (_Float16)bp1[j];
      *(unsigned*)&Bsh[bc0 + j][2 * kp] = pk.u;
    }
    if (k0 < 480) __builtin_prefetch(bp0 + 32 * 512);
    asm volatile("s_wait_asynccnt 0" ::: "memory");
    __syncthreads();
    // batch all frag loads -> one DS drain -> 8 consecutive WMMAs
    v16h a = ld2(&Ash[wv * 16 + n][g * 8], &Ash[wv * 16 + n][16 + g * 8]);
    v16h bf[8];
    #pragma unroll
    for (int nt = 0; nt < 8; nt++)
      bf[nt] = ld2(&Bsh[nt * 16 + n][g * 8], &Bsh[nt * 16 + n][16 + g * 8]);
    #pragma unroll
    for (int nt = 0; nt < 8; nt++)
      acc[nt] = WMMA_F16(a, bf[nt], acc[nt]);
  }

  #pragma unroll
  for (int nt = 0; nt < 8; nt++) {
    int gcol = n0 + nt * 16 + n;
    float bb = bout[gcol];
    #pragma unroll
    for (int r = 0; r < 8; r++) {
      int row = m0 + wv * 16 + r + 8 * g;
      int b = row >> 12, yy = (row >> 6) & 63, xx = row & 63;
      out[(((size_t)b * 64 + ((yy + 4) & 63)) * 64 + ((xx + 4) & 63)) * 512 + gcol] =
          acc[nt][r] + bb;
    }
  }
}

// ---------------------------------------------------------------------------
extern "C" void kernel_launch(void* const* d_in, const int* in_sizes, int n_in,
                              void* d_out, int out_size, void* d_ws, size_t ws_size,
                              hipStream_t stream) {
  const float* x    = (const float*)d_in[0];
  const float* wqkv = (const float*)d_in[1];
  const float* ls   = (const float*)d_in[2];
  const float* cw1  = (const float*)d_in[3];
  const float* cb1  = (const float*)d_in[4];
  const float* cw2  = (const float*)d_in[5];
  const float* cb2  = (const float*)d_in[6];
  const float* wout = (const float*)d_in[7];
  const float* bout = (const float*)d_in[8];
  float* out = (float*)d_out;

  // workspace layout (~80.3 MB):
  float* qw = (float*)d_ws;                           // 4,194,304 f32
  float* kw = qw + (size_t)4194304;                   // 4,194,304 f32
  float* vw = kw + (size_t)4194304;                   // 4,194,304 f32
  _Float16* att = (_Float16*)(vw + (size_t)4194304);  // 16,777,216 f16
  float* biasw = (float*)(att + (size_t)16777216);    // 65,536 f32

  cpb_bias_kernel<<<16, 256, 0, stream>>>(cw1, cb1, cw2, cb2, biasw);
  gemm_qkv_kernel<<<dim3(512, 12), 128, 0, stream>>>(x, wqkv, qw, kw, vw);
  attn_kernel<<<8192, 128, 0, stream>>>(qw, kw, vw, ls, biasw, att);
  gemm_out_kernel<<<dim3(512, 4), 128, 0, stream>>>(att, wout, bout, out);
}
